// TrueEpisodicMemory_27797028339950
// MI455X (gfx1250) — matrix operand order repaired and checked by reading, not compiled, not run on hardware
//
#include <hip/hip_runtime.h>
#include <hip/hip_bf16.h>

typedef __attribute__((ext_vector_type(16))) __bf16 v16bf;
typedef __attribute__((ext_vector_type(8)))  float  v8f;

#define B_TOTAL 131072
#define M_SLOTS 100
#define MP      112          // slots padded to 7*16
#define D_DIM   1024
#define C_DIM   16
#define KX      1056         // 1024 + 32 (context fold + pad), multiple of 32
#define KA      128          // attn K padded to 4*32
#define TIME_W  0.1f

#define KEYSX_ELEMS (MP * KX)       // 118272 bf16 = 236544 B
#define VALST_ELEMS (D_DIM * KA)    // 131072 bf16 = 262144 B
#define BIG_ELEMS   VALST_ELEMS     // max of the two

// ---------------- workspace layout ----------------
#define WS_KEYSX 0
#define WS_VALST 236544
#define WS_BIAS  (236544 + 262144)

__global__ __launch_bounds__(256)
void epis_prep(const float* __restrict__ keys, const float* __restrict__ vals,
               const float* __restrict__ ctxs, const float* __restrict__ ts,
               __bf16* __restrict__ keysX, __bf16* __restrict__ valsT,
               float* __restrict__ bias)
{
    const int tid  = blockIdx.x * blockDim.x + threadIdx.x;
    const int nthr = gridDim.x * blockDim.x;

    for (int idx = tid; idx < MP * KX; idx += nthr) {
        int m = idx / KX, k = idx - m * KX;
        float v = 0.0f;
        if (m < M_SLOTS) {
            if (k < D_DIM)              v = keys[m * D_DIM + k];
            else if (k < D_DIM + C_DIM) v = 0.5f * ctxs[m * C_DIM + (k - D_DIM)];
        }
        keysX[idx] = (__bf16)v;
    }
    for (int idx = tid; idx < D_DIM * KA; idx += nthr) {
        int d = idx >> 7, m = idx & (KA - 1);
        float v = (m < M_SLOTS) ? vals[m * D_DIM + d] : 0.0f;
        valsT[idx] = (__bf16)v;
    }
    for (int idx = tid; idx < MP; idx += nthr) {
        bias[idx] = (idx < M_SLOTS) ? __expf(TIME_W * ts[idx]) : -1e30f;
    }
}

__global__ __launch_bounds__(256)
void epis_main(const float* __restrict__ q, const float* __restrict__ ctx,
               const __bf16* __restrict__ keysX, const __bf16* __restrict__ valsT,
               const float* __restrict__ bias, float* __restrict__ out)
{
    // 256 KB: holds keysX during GEMM1, restaged with valsT for GEMM2
    __shared__ __bf16 bigLds[BIG_ELEMS];
    // 32 KB: per-wave attn D->A re-layout tile (reused for both M-tiles)
    __shared__ __bf16 attnLds[8][16][KA];

    const int tid  = threadIdx.x;
    const int lane = tid & 31;
    const int wave = tid >> 5;
    const int half = lane >> 4;      // K/M half-wave split (ISA 7.12.2)
    const int ln   = lane & 15;      // N (or A-row) index within tile
    const int row0 = blockIdx.x * 256 + wave * 32;   // 2 M-tiles of 16 rows per wave

    // ---- stage keysX into LDS (cooperative, b128 chunks) ----
    {
        const uint4* src = (const uint4*)keysX;
        uint4*       dst = (uint4*)bigLds;
        for (int i = tid; i < KEYSX_ELEMS / 8; i += 256) dst[i] = src[i];
    }
    __syncthreads();

    // ---------------- GEMM1: scores[2][16 x 112] ----------------
    v8f acc[2][7] = {};

    const float* qrow0 = q   + (size_t)(row0 + ln) * D_DIM;
    const float* crow0 = ctx + (size_t)(row0 + ln) * C_DIM;
    const float* qrow1 = qrow0 + (size_t)16 * D_DIM;
    const float* crow1 = crow0 + (size_t)16 * C_DIM;

    auto loadA = [&](const float* qr, const float* cr, int ks) -> v16bf {
        v16bf a;
        const int kb = ks * 32 + half * 8;
        if (ks < 32) {
            const float4 x0 = *(const float4*)(qr + kb);
            const float4 x1 = *(const float4*)(qr + kb + 4);
            const float4 x2 = *(const float4*)(qr + kb + 16);
            const float4 x3 = *(const float4*)(qr + kb + 20);
            a[0]=(__bf16)x0.x; a[1]=(__bf16)x0.y; a[2]=(__bf16)x0.z; a[3]=(__bf16)x0.w;
            a[4]=(__bf16)x1.x; a[5]=(__bf16)x1.y; a[6]=(__bf16)x1.z; a[7]=(__bf16)x1.w;
            a[8]=(__bf16)x2.x; a[9]=(__bf16)x2.y; a[10]=(__bf16)x2.z; a[11]=(__bf16)x2.w;
            a[12]=(__bf16)x3.x; a[13]=(__bf16)x3.y; a[14]=(__bf16)x3.z; a[15]=(__bf16)x3.w;
        } else {
            // last K-step: cols 1024..1039 = context, 1040..1055 = 0
            const float4 c0 = *(const float4*)(cr + half * 8);
            const float4 c1 = *(const float4*)(cr + half * 8 + 4);
            a[0]=(__bf16)c0.x; a[1]=(__bf16)c0.y; a[2]=(__bf16)c0.z; a[3]=(__bf16)c0.w;
            a[4]=(__bf16)c1.x; a[5]=(__bf16)c1.y; a[6]=(__bf16)c1.z; a[7]=(__bf16)c1.w;
            #pragma unroll
            for (int j = 8; j < 16; ++j) a[j] = (__bf16)0.0f;
        }
        return a;
    };

    for (int ks = 0; ks < 33; ++ks) {
        const int k0 = ks * 32;
        v16bf a0 = loadA(qrow0, crow0, ks);
        v16bf a1 = loadA(qrow1, crow1, ks);
        #pragma unroll
        for (int t = 0; t < 7; ++t) {
            // B fragment from LDS: slot n = t*16+ln, K = 8h+j and 16+8h+j
            const __bf16* bp = bigLds + (size_t)(t * 16 + ln) * KX + k0 + half * 8;
            v16bf b;
            #pragma unroll
            for (int j = 0; j < 8; ++j) { b[j] = bp[j]; b[8 + j] = bp[16 + j]; }
            acc[0][t] = __builtin_amdgcn_wmma_f32_16x16x32_bf16(
                            false, a0, false, b, (short)0, acc[0][t], false, false);
            acc[1][t] = __builtin_amdgcn_wmma_f32_16x16x32_bf16(
                            false, a1, false, b, (short)0, acc[1][t], false, false);
        }
    }

    // ---------------- bias + softmax (registers only) ----------------
    #pragma unroll
    for (int mt = 0; mt < 2; ++mt) {
        #pragma unroll
        for (int t = 0; t < 7; ++t) {
            const float bc = bias[t * 16 + ln];
            #pragma unroll
            for (int i = 0; i < 8; ++i) acc[mt][t][i] += bc;
        }
        #pragma unroll
        for (int i = 0; i < 8; ++i) {
            float mx = acc[mt][0][i];
            #pragma unroll
            for (int t = 1; t < 7; ++t) mx = fmaxf(mx, acc[mt][t][i]);
            #pragma unroll
            for (int s = 1; s < 16; s <<= 1) mx = fmaxf(mx, __shfl_xor(mx, s, 32));
            float e[7], sum = 0.0f;
            #pragma unroll
            for (int t = 0; t < 7; ++t) { e[t] = __expf(acc[mt][t][i] - mx); sum += e[t]; }
            #pragma unroll
            for (int s = 1; s < 16; s <<= 1) sum += __shfl_xor(sum, s, 32);
            const float r = __frcp_rn(sum);
            #pragma unroll
            for (int t = 0; t < 7; ++t) acc[mt][t][i] = e[t] * r;
        }
    }

    __syncthreads();                       // all waves done reading keysX from LDS

    // ---- restage LDS with valsT (overlaps with attn re-layout below) ----
    {
        const uint4* src = (const uint4*)valsT;
        uint4*       dst = (uint4*)bigLds;
        for (int i = tid; i < VALST_ELEMS / 8; i += 256) dst[i] = src[i];
    }

    // ---- attn D-layout -> A-layout via per-wave LDS tile; preload A fragments ----
    v16bf af[2][4];
    #pragma unroll
    for (int mt = 0; mt < 2; ++mt) {
        #pragma unroll
        for (int t = 0; t < 7; ++t)
            #pragma unroll
            for (int i = 0; i < 8; ++i)
                attnLds[wave][i + half * 8][t * 16 + ln] = (__bf16)acc[mt][t][i];
        #pragma unroll
        for (int i = 0; i < 8; ++i)
            attnLds[wave][i + half * 8][112 + ln] = (__bf16)0.0f;   // pad K to 128
        #pragma unroll
        for (int ks = 0; ks < 4; ++ks) {
            const int kb = ks * 32 + half * 8;
            #pragma unroll
            for (int j = 0; j < 8; ++j) {
                af[mt][ks][j]     = attnLds[wave][ln][kb + j];
                af[mt][ks][8 + j] = attnLds[wave][ln][kb + 16 + j];
            }
        }
    }

    __syncthreads();                       // valsT fully staged

    // ---------------- GEMM2: out[2][16 x 1024] = attn * values ----------------
    for (int nt = 0; nt < 64; ++nt) {
        const int n0 = nt * 16;
        v8f c0 = {}, c1 = {};
        #pragma unroll
        for (int ks = 0; ks < 4; ++ks) {
            const __bf16* bp = bigLds + (size_t)(n0 + ln) * KA + ks * 32 + half * 8;
            v16bf b;
            #pragma unroll
            for (int j = 0; j < 8; ++j) { b[j] = bp[j]; b[8 + j] = bp[16 + j]; }
            c0 = __builtin_amdgcn_wmma_f32_16x16x32_bf16(
                     false, af[0][ks], false, b, (short)0, c0, false, false);
            c1 = __builtin_amdgcn_wmma_f32_16x16x32_bf16(
                     false, af[1][ks], false, b, (short)0, c1, false, false);
        }
        #pragma unroll
        for (int i = 0; i < 8; ++i) {
            out[(size_t)(row0 + i + half * 8) * D_DIM + n0 + ln]      = c0[i];
            out[(size_t)(row0 + 16 + i + half * 8) * D_DIM + n0 + ln] = c1[i];
        }
    }
}

extern "C" void kernel_launch(void* const* d_in, const int* in_sizes, int n_in,
                              void* d_out, int out_size, void* d_ws, size_t ws_size,
                              hipStream_t stream) {
    const float* query    = (const float*)d_in[0];
    const float* context  = (const float*)d_in[1];
    const float* mem_keys = (const float*)d_in[2];
    const float* mem_vals = (const float*)d_in[3];
    const float* mem_ctxs = (const float*)d_in[4];
    const float* mem_ts   = (const float*)d_in[5];
    float* out = (float*)d_out;

    char* ws = (char*)d_ws;
    __bf16* keysX = (__bf16*)(ws + WS_KEYSX);
    __bf16* valsT = (__bf16*)(ws + WS_VALST);
    float*  bias  = (float*)(ws + WS_BIAS);

    epis_prep<<<256, 256, 0, stream>>>(mem_keys, mem_vals, mem_ctxs, mem_ts,
                                       keysX, valsT, bias);
    epis_main<<<B_TOTAL / 256, 256, 0, stream>>>(query, context, keysX, valsT,
                                                 bias, out);
}
